// FlattenSELayer_31834297598792
// MI455X (gfx1250) — matrix operand order repaired and checked by reading, not compiled, not run on hardware
//
#include <hip/hip_runtime.h>

typedef _Float16 v16h __attribute__((ext_vector_type(16)));
typedef float    v8f  __attribute__((ext_vector_type(8)));

#define NSEG 16
#define CH   256
#define HID  64

// ---------------------------------------------------------------------------
// Kernel 0: zero the accumulator region of the workspace (4096 sums + 16 cnts)
// ---------------------------------------------------------------------------
__global__ void zero_ws_kernel(float* __restrict__ ws, int n) {
    int i = blockIdx.x * blockDim.x + threadIdx.x;
    if (i < n) ws[i] = 0.0f;
}

// ---------------------------------------------------------------------------
// Kernel 1: segment sum + counts.  LDS accumulators per block (16KB),
// ds_add_f32 atomics (conflict-free lane->channel mapping), float4 loads,
// 2-row unroll for memory-level parallelism, global f32 atomics to finalize.
// ---------------------------------------------------------------------------
__global__ __launch_bounds__(256) void seg_reduce_kernel(
    const float* __restrict__ x, const int* __restrict__ idx,
    float* __restrict__ seg_sum, float* __restrict__ seg_cnt, int nrows) {
    __shared__ float lsum[NSEG * CH];
    __shared__ float lcnt[NSEG];
    const int tid = threadIdx.x;
    for (int i = tid; i < NSEG * CH; i += 256) lsum[i] = 0.0f;
    if (tid < NSEG) lcnt[tid] = 0.0f;
    __syncthreads();

    const float4* x4 = reinterpret_cast<const float4*>(x);
    const int lane = tid & 63;   // float4 column 0..63
    const int rsub = tid >> 6;   // sub-row 0..3
    const int stride = gridDim.x * 8;
    for (int row = blockIdx.x * 8 + rsub; row < nrows; row += stride) {
        const int  row2 = row + 4;
        const bool h2   = row2 < nrows;
        const int  seg1 = idx[row];
        const float4 v1 = x4[row * 64 + lane];
        int seg2 = 0;
        float4 v2;
        if (h2) { seg2 = idx[row2]; v2 = x4[row2 * 64 + lane]; }

        const int b1 = seg1 * CH + lane * 4;
        atomicAdd(&lsum[b1 + 0], v1.x);
        atomicAdd(&lsum[b1 + 1], v1.y);
        atomicAdd(&lsum[b1 + 2], v1.z);
        atomicAdd(&lsum[b1 + 3], v1.w);
        if (lane == 0) atomicAdd(&lcnt[seg1], 1.0f);
        if (h2) {
            const int b2 = seg2 * CH + lane * 4;
            atomicAdd(&lsum[b2 + 0], v2.x);
            atomicAdd(&lsum[b2 + 1], v2.y);
            atomicAdd(&lsum[b2 + 2], v2.z);
            atomicAdd(&lsum[b2 + 3], v2.w);
            if (lane == 0) atomicAdd(&lcnt[seg2], 1.0f);
        }
    }
    __syncthreads();
    for (int i = tid; i < NSEG * CH; i += 256) atomicAdd(&seg_sum[i], lsum[i]);
    if (tid < NSEG) atomicAdd(&seg_cnt[tid], lcnt[tid]);
}

// ---------------------------------------------------------------------------
// Kernel 2: SE MLP on one wave (EXEC all-1s, wave32) with WMMA f16->f32.
//   h    = relu(mean @ w1)      M=16 K=256 N=64   -> 4 N-tiles x 8 K-steps
//   gate = sigmoid(h @ w2)      M=16 K=64  N=256  -> 16 N-tiles x 2 K-steps
// Fragment layouts per CDNA5 ISA 7.12.2:
//   A (16x32 f16): lane m (0..15) h[0..7]=K0..7,h[8..15]=K16..23; lane m+16
//                  holds K8..15 / K24..31.
//   B (32x16 f16): lane = N col; halfs 0..7 = K0..7, 8..15 = K16..23 (lanes 16..31: K+8).
//   C/D (16x16 f32): VGPR r -> M = 8*(lane>=16)+r, N = lane&15.
// ---------------------------------------------------------------------------
__global__ __launch_bounds__(32) void se_mlp_kernel(
    const float* __restrict__ seg_sum, const float* __restrict__ seg_cnt,
    const float* __restrict__ w1, const float* __restrict__ w2,
    float* __restrict__ gate) {
    __shared__ _Float16 meanh[NSEG * CH];   // 16x256 segment means (f16)
    __shared__ _Float16 hmat[NSEG * HID];   // 16x64 hidden activations (f16)

    const int l  = threadIdx.x;   // 0..31
    const int m  = l & 15;        // row / column within tile
    const int kh = l >> 4;        // lane-half selector

    // segment mean = seg_sum / max(count, 1)
    for (int i = l; i < NSEG * CH; i += 32) {
        float c = seg_cnt[i >> 8];
        c = c > 1.0f ? c : 1.0f;
        meanh[i] = (_Float16)(seg_sum[i] / c);
    }
    __syncthreads();

    // ---- GEMM1: h = relu(mean @ w1) ----
    for (int nt = 0; nt < 4; ++nt) {
        v8f acc = {};
#pragma unroll
        for (int kt = 0; kt < 8; ++kt) {
            const int kbase = kt * 32;
            v16h a, b;
#pragma unroll
            for (int j = 0; j < 8; ++j) {
                a[j]     = meanh[m * CH + kbase + kh * 8 + j];
                a[8 + j] = meanh[m * CH + kbase + 16 + kh * 8 + j];
            }
#pragma unroll
            for (int j = 0; j < 16; ++j) {
                const int k = kbase + kh * 8 + (j < 8 ? j : 8 + j);
                b[j] = (_Float16)w1[k * HID + nt * 16 + m];
            }
            acc = __builtin_amdgcn_wmma_f32_16x16x32_f16(
                false, a, false, b, (short)0, acc, false, false);
        }
#pragma unroll
        for (int r = 0; r < 8; ++r) {
            float v = acc[r] > 0.0f ? acc[r] : 0.0f;   // ReLU
            hmat[(kh * 8 + r) * HID + nt * 16 + m] = (_Float16)v;
        }
    }
    __syncthreads();

    // ---- GEMM2: gate = sigmoid(h @ w2) ----
    for (int nt = 0; nt < 16; ++nt) {
        v8f acc = {};
#pragma unroll
        for (int kt = 0; kt < 2; ++kt) {
            const int kbase = kt * 32;
            v16h a, b;
#pragma unroll
            for (int j = 0; j < 8; ++j) {
                a[j]     = hmat[m * HID + kbase + kh * 8 + j];
                a[8 + j] = hmat[m * HID + kbase + 16 + kh * 8 + j];
            }
#pragma unroll
            for (int j = 0; j < 16; ++j) {
                const int k = kbase + kh * 8 + (j < 8 ? j : 8 + j);
                b[j] = (_Float16)w2[k * CH + nt * 16 + m];
            }
            acc = __builtin_amdgcn_wmma_f32_16x16x32_f16(
                false, a, false, b, (short)0, acc, false, false);
        }
#pragma unroll
        for (int r = 0; r < 8; ++r) {
            float g = 1.0f / (1.0f + __expf(-acc[r]));  // sigmoid
            gate[(kh * 8 + r) * CH + nt * 16 + m] = g;
        }
    }
}

// ---------------------------------------------------------------------------
// Kernel 3: out[i,:] = x[i,:] * gate[idx[i],:].  gate cached in LDS (16KB),
// float4 traffic, 2-row unroll.
// ---------------------------------------------------------------------------
__global__ __launch_bounds__(256) void modulate_kernel(
    const float* __restrict__ x, const int* __restrict__ idx,
    const float* __restrict__ gate, float* __restrict__ out, int nrows) {
    __shared__ float4 g4s[NSEG * 64];
    const int tid = threadIdx.x;
    const float4* g4 = reinterpret_cast<const float4*>(gate);
    for (int i = tid; i < NSEG * 64; i += 256) g4s[i] = g4[i];
    __syncthreads();

    const float4* x4 = reinterpret_cast<const float4*>(x);
    float4* o4 = reinterpret_cast<float4*>(out);
    const int lane = tid & 63;
    const int rsub = tid >> 6;
    const int stride = gridDim.x * 8;
    for (int row = blockIdx.x * 8 + rsub; row < nrows; row += stride) {
        const int  row2 = row + 4;
        const bool h2   = row2 < nrows;
        const int  seg1 = idx[row];
        const float4 v1 = x4[row * 64 + lane];
        int seg2 = 0;
        float4 v2;
        if (h2) { seg2 = idx[row2]; v2 = x4[row2 * 64 + lane]; }

        const float4 g1 = g4s[seg1 * 64 + lane];
        o4[row * 64 + lane] =
            make_float4(v1.x * g1.x, v1.y * g1.y, v1.z * g1.z, v1.w * g1.w);
        if (h2) {
            const float4 g2 = g4s[seg2 * 64 + lane];
            o4[row2 * 64 + lane] =
                make_float4(v2.x * g2.x, v2.y * g2.y, v2.z * g2.z, v2.w * g2.w);
        }
    }
}

// ---------------------------------------------------------------------------
extern "C" void kernel_launch(void* const* d_in, const int* in_sizes, int n_in,
                              void* d_out, int out_size, void* d_ws, size_t ws_size,
                              hipStream_t stream) {
    const float* x   = (const float*)d_in[0];
    const int*   idx = (const int*)  d_in[1];
    const float* w1  = (const float*)d_in[2];
    const float* w2  = (const float*)d_in[3];
    float* out = (float*)d_out;
    float* ws  = (float*)d_ws;

    float* seg_sum = ws;              // 4096 floats
    float* seg_cnt = ws + 4096;       // 16 floats
    float* gate    = ws + 4112;       // 4096 floats (16B-aligned offset)

    const int nrows = in_sizes[0] / CH;

    zero_ws_kernel<<<(4112 + 255) / 256, 256, 0, stream>>>(ws, 4112);
    seg_reduce_kernel<<<1024, 256, 0, stream>>>(x, idx, seg_sum, seg_cnt, nrows);
    se_mlp_kernel<<<1, 32, 0, stream>>>(seg_sum, seg_cnt, w1, w2, gate);
    modulate_kernel<<<2048, 256, 0, stream>>>(x, idx, gate, out, nrows);
}
